// CausalSelfAttention_55147380081061
// MI455X (gfx1250) — matrix operand order repaired and checked by reading, not compiled, not run on hardware
//
#include <hip/hip_runtime.h>
#include <math.h>

// Problem constants (fixed by the reference)
#define Bv 4
#define Tv 2048
#define Cv 1024
#define Hv 16
#define Dv 64

typedef __attribute__((ext_vector_type(16))) __bf16 v16bf;
typedef __attribute__((ext_vector_type(8)))  float  v8f;
typedef __attribute__((ext_vector_type(4)))  unsigned int u32x4;

union ABFrag { v16bf v; u32x4 q[2]; };

static __device__ __forceinline__ __bf16 f2bf(float f) {
  union { float f; unsigned u; } a; a.f = f;
  unsigned r = a.u + 0x7FFFu + ((a.u >> 16) & 1u);  // round-to-nearest-even
  unsigned short h = (unsigned short)(r >> 16);
  union { unsigned short s; __bf16 b; } c; c.s = h;
  return c.b;
}

static __device__ __forceinline__ v8f wmma_bf16(v16bf a, v16bf b, v8f c) {
  return __builtin_amdgcn_wmma_f32_16x16x32_bf16(false, a, false, b, (short)0, c,
                                                 false, false);
}

// A-matrix fragment 16(M)x32(K) bf16, row-major source with stride lda.
// Per ISA 7.12.2: lane L holds row m=L&15; K groups {half*8..+7} and {16+half*8..+7}.
static __device__ __forceinline__ v16bf load_a16x32(const __bf16* base, int lda) {
  const int lane = threadIdx.x & 31;
  const int m  = lane & 15;
  const int kb = (lane >> 4) << 3;
  const __bf16* p = base + m * lda;
  ABFrag f;
  f.q[0] = *reinterpret_cast<const u32x4*>(p + kb);
  f.q[1] = *reinterpret_cast<const u32x4*>(p + 16 + kb);
  return f.v;
}

// B-matrix fragment 32(K)x16(N) bf16 from a "weight-style" [n][k] row-major
// source with stride ldb. Lane L holds column n=L&15, K groups {half*8..+7} and {16+half*8..+7}.
static __device__ __forceinline__ v16bf load_b32x16(const __bf16* base, int ldb) {
  const int lane = threadIdx.x & 31;
  const int n  = lane & 15;
  const int kb = (lane >> 4) << 3;
  const __bf16* p = base + n * ldb + kb;
  ABFrag f;
  f.q[0] = *reinterpret_cast<const u32x4*>(p);
  f.q[1] = *reinterpret_cast<const u32x4*>(p + 16);
  return f.v;
}

// ---------------------------------------------------------------- convert
__global__ void cvt_f32_bf16(const float* __restrict__ in,
                             __bf16* __restrict__ out, int n4) {
  int i = blockIdx.x * blockDim.x + threadIdx.x;
  const int stride = gridDim.x * blockDim.x;
  for (; i < n4; i += stride) {
    float4 v = reinterpret_cast<const float4*>(in)[i];
    union { __bf16 b[4]; unsigned long long u; } pk;
    pk.b[0] = f2bf(v.x); pk.b[1] = f2bf(v.y);
    pk.b[2] = f2bf(v.z); pk.b[3] = f2bf(v.w);
    reinterpret_cast<unsigned long long*>(out)[i] = pk.u;
  }
}

// ---------------------------------------------------------------- QKV GEMM
// qkv[m][o] = sum_c X[m][c] * Wqkv[o][c]; scatter into Q,K ([b][h][t][d]) and
// V transposed ([b][h][d][t]).  8 waves/block: 4 (M) x 2 (N), 64x64 per wave.
__global__ __launch_bounds__(256)
void qkv_gemm(const __bf16* __restrict__ X, const __bf16* __restrict__ W,
              __bf16* __restrict__ Qb, __bf16* __restrict__ Kb,
              __bf16* __restrict__ VT) {
  const int lane = threadIdx.x & 31;
  const int wave = threadIdx.x >> 5;
  const int mw = blockIdx.x * 256 + (wave >> 1) * 64;
  const int nw = blockIdx.y * 128 + (wave & 1) * 64;
  const int K = Cv;

  v8f acc[4][4];
#pragma unroll
  for (int i = 0; i < 4; ++i)
#pragma unroll
    for (int j = 0; j < 4; ++j)
#pragma unroll
      for (int e = 0; e < 8; ++e) acc[i][j][e] = 0.0f;

  for (int kk = 0; kk < K; kk += 32) {
    v16bf a[4], b[4];
#pragma unroll
    for (int i = 0; i < 4; ++i) a[i] = load_a16x32(X + (size_t)(mw + 16 * i) * K + kk, K);
#pragma unroll
    for (int j = 0; j < 4; ++j) b[j] = load_b32x16(W + (size_t)(nw + 16 * j) * K + kk, K);
#pragma unroll
    for (int i = 0; i < 4; ++i)
#pragma unroll
      for (int j = 0; j < 4; ++j) acc[i][j] = wmma_bf16(a[i], b[j], acc[i][j]);
  }

  const int half8 = (lane >> 4) << 3;
  const int nl = lane & 15;
#pragma unroll
  for (int i = 0; i < 4; ++i)
#pragma unroll
    for (int j = 0; j < 4; ++j) {
      const int o = nw + 16 * j + nl;          // [0, 3C)
      const int sec = o >> 10;                 // 0=q 1=k 2=v
      const int oc = o & (Cv - 1);
      const int h = oc >> 6, d = oc & (Dv - 1);
#pragma unroll
      for (int r = 0; r < 8; ++r) {
        const int mg = mw + 16 * i + half8 + r;   // [0, B*T)
        const int bi = mg >> 11, t = mg & (Tv - 1);
        const int bh = bi * Hv + h;
        const __bf16 val = f2bf(acc[i][j][r]);
        if (sec == 0)      Qb[((size_t)bh * Tv + t) * Dv + d] = val;
        else if (sec == 1) Kb[((size_t)bh * Tv + t) * Dv + d] = val;
        else               VT[((size_t)bh * Dv + d) * Tv + t] = val;
      }
    }
}

// ---------------------------------------------------------------- attention
// One wave owns 32 query rows of one (b,h); loops key blocks of 32 with online
// softmax (flash).  S = Q·K^T via WMMA, P·V via WMMA with P routed through LDS
// to switch C-layout -> A-layout.
__global__ __launch_bounds__(256)
void flash_attn(const __bf16* __restrict__ Qb, const __bf16* __restrict__ Kb,
                const __bf16* __restrict__ VT, __bf16* __restrict__ O) {
  __shared__ __align__(16) __bf16 psh[8][32 * 32];

  const int lane = threadIdx.x & 31;
  const int wave = threadIdx.x >> 5;
  const int bh = blockIdx.x >> 3;
  const int tile = ((blockIdx.x & 7) << 3) + wave;  // 0..63
  const int q0 = tile * 32;
  const int bi = bh >> 4, h = bh & 15;

  const __bf16* q  = Qb + (size_t)bh * Tv * Dv;
  const __bf16* k  = Kb + (size_t)bh * Tv * Dv;
  const __bf16* vt = VT + (size_t)bh * Dv * Tv;

  const int half8 = (lane >> 4) << 3;
  const int nl = lane & 15;

  v8f o_acc[2][4];
  float m_run[2][8], l_run[2][8];
#pragma unroll
  for (int i = 0; i < 2; ++i) {
#pragma unroll
    for (int j = 0; j < 4; ++j)
#pragma unroll
      for (int e = 0; e < 8; ++e) o_acc[i][j][e] = 0.0f;
#pragma unroll
    for (int r = 0; r < 8; ++r) { m_run[i][r] = -__builtin_inff(); l_run[i][r] = 0.0f; }
  }

  for (int kt = 0; kt < q0 + 32; kt += 32) {
    // ---- S = (Q K^T) * 1/sqrt(D), 64 contraction in 2 steps of 32
    v8f s[2][2];
#pragma unroll
    for (int i = 0; i < 2; ++i)
#pragma unroll
      for (int j = 0; j < 2; ++j)
#pragma unroll
        for (int e = 0; e < 8; ++e) s[i][j][e] = 0.0f;

#pragma unroll
    for (int ks = 0; ks < Dv; ks += 32) {
      v16bf aq[2], bk[2];
#pragma unroll
      for (int i = 0; i < 2; ++i) aq[i] = load_a16x32(q + (size_t)(q0 + 16 * i) * Dv + ks, Dv);
#pragma unroll
      for (int j = 0; j < 2; ++j) bk[j] = load_b32x16(k + (size_t)(kt + 16 * j) * Dv + ks, Dv);
#pragma unroll
      for (int i = 0; i < 2; ++i)
#pragma unroll
        for (int j = 0; j < 2; ++j) s[i][j] = wmma_bf16(aq[i], bk[j], s[i][j]);
    }

    // ---- scale + causal mask
#pragma unroll
    for (int i = 0; i < 2; ++i)
#pragma unroll
      for (int j = 0; j < 2; ++j)
#pragma unroll
        for (int r = 0; r < 8; ++r) {
          const int mrow = q0 + 16 * i + half8 + r;
          const int ncol = kt + 16 * j + nl;
          const float sv = s[i][j][r] * 0.125f;
          s[i][j][r] = (ncol > mrow) ? -__builtin_inff() : sv;
        }

    // ---- online softmax per M-tile; write P (bf16) to this wave's LDS tile
#pragma unroll
    for (int i = 0; i < 2; ++i) {
      float mx[8], sm[8], alpha[8], mnew[8];
#pragma unroll
      for (int r = 0; r < 8; ++r) mx[r] = fmaxf(s[i][0][r], s[i][1][r]);
#pragma unroll
      for (int msk = 1; msk < 16; msk <<= 1)
#pragma unroll
        for (int r = 0; r < 8; ++r) mx[r] = fmaxf(mx[r], __shfl_xor(mx[r], msk));
#pragma unroll
      for (int r = 0; r < 8; ++r) {
        mnew[r] = fmaxf(m_run[i][r], mx[r]);
        alpha[r] = __expf(m_run[i][r] - mnew[r]);
        m_run[i][r] = mnew[r];
        sm[r] = 0.0f;
      }
#pragma unroll
      for (int j = 0; j < 2; ++j)
#pragma unroll
        for (int r = 0; r < 8; ++r) {
          const float p = __expf(s[i][j][r] - mnew[r]);   // masked -> 0
          sm[r] += p;
          psh[wave][(16 * i + half8 + r) * 32 + 16 * j + nl] = f2bf(p);
        }
#pragma unroll
      for (int msk = 1; msk < 16; msk <<= 1)
#pragma unroll
        for (int r = 0; r < 8; ++r) sm[r] += __shfl_xor(sm[r], msk);
#pragma unroll
      for (int r = 0; r < 8; ++r) l_run[i][r] = l_run[i][r] * alpha[r] + sm[r];
#pragma unroll
      for (int j = 0; j < 4; ++j)
#pragma unroll
        for (int r = 0; r < 8; ++r) o_acc[i][j][r] *= alpha[r];
    }

    asm volatile("s_wait_dscnt 0" ::: "memory");  // P stores visible to this wave

    // ---- O += P · V   (P: A-frag from LDS, V^T: contiguous B-frags)
    v16bf pa[2];
#pragma unroll
    for (int i = 0; i < 2; ++i) {
      const __bf16* p = &psh[wave][(size_t)(16 * i + nl) * 32];
      ABFrag f;
      f.q[0] = *reinterpret_cast<const u32x4*>(p + half8);
      f.q[1] = *reinterpret_cast<const u32x4*>(p + 16 + half8);
      pa[i] = f.v;
    }
#pragma unroll
    for (int j = 0; j < 4; ++j) {
      v16bf bv = load_b32x16(vt + (size_t)(16 * j) * Tv + kt, Tv);
#pragma unroll
      for (int i = 0; i < 2; ++i) o_acc[i][j] = wmma_bf16(pa[i], bv, o_acc[i][j]);
    }
  }

  // ---- normalize + write O in [b][t][h*D+d] bf16 layout for the proj GEMM
#pragma unroll
  for (int i = 0; i < 2; ++i)
#pragma unroll
    for (int j = 0; j < 4; ++j) {
      const int d = 16 * j + nl;
#pragma unroll
      for (int r = 0; r < 8; ++r) {
        const int t = q0 + 16 * i + half8 + r;
        const float val = o_acc[i][j][r] / l_run[i][r];
        O[((size_t)(bi * Tv + t)) * Cv + h * Dv + d] = f2bf(val);
      }
    }
}

// ---------------------------------------------------------------- out proj
__global__ __launch_bounds__(256)
void proj_gemm(const __bf16* __restrict__ A, const __bf16* __restrict__ W,
               float* __restrict__ out) {
  const int lane = threadIdx.x & 31;
  const int wave = threadIdx.x >> 5;
  const int mw = blockIdx.x * 256 + (wave >> 1) * 64;
  const int nw = blockIdx.y * 128 + (wave & 1) * 64;
  const int K = Cv;

  v8f acc[4][4];
#pragma unroll
  for (int i = 0; i < 4; ++i)
#pragma unroll
    for (int j = 0; j < 4; ++j)
#pragma unroll
      for (int e = 0; e < 8; ++e) acc[i][j][e] = 0.0f;

  for (int kk = 0; kk < K; kk += 32) {
    v16bf a[4], b[4];
#pragma unroll
    for (int i = 0; i < 4; ++i) a[i] = load_a16x32(A + (size_t)(mw + 16 * i) * K + kk, K);
#pragma unroll
    for (int j = 0; j < 4; ++j) b[j] = load_b32x16(W + (size_t)(nw + 16 * j) * K + kk, K);
#pragma unroll
    for (int i = 0; i < 4; ++i)
#pragma unroll
      for (int j = 0; j < 4; ++j) acc[i][j] = wmma_bf16(a[i], b[j], acc[i][j]);
  }

  const int half8 = (lane >> 4) << 3;
  const int nl = lane & 15;
#pragma unroll
  for (int i = 0; i < 4; ++i)
#pragma unroll
    for (int j = 0; j < 4; ++j) {
      const int n = nw + 16 * j + nl;
#pragma unroll
      for (int r = 0; r < 8; ++r) {
        const int mg = mw + 16 * i + half8 + r;
        out[(size_t)mg * Cv + n] = acc[i][j][r];
      }
    }
}

// ---------------------------------------------------------------- launcher
extern "C" void kernel_launch(void* const* d_in, const int* in_sizes, int n_in,
                              void* d_out, int out_size, void* d_ws, size_t ws_size,
                              hipStream_t stream) {
  (void)in_sizes; (void)n_in; (void)out_size; (void)ws_size;
  const float* x     = (const float*)d_in[0];
  const float* wqkv  = (const float*)d_in[1];
  const float* wproj = (const float*)d_in[2];
  float* out = (float*)d_out;

  char* ws = (char*)d_ws;
  size_t off = 0;
  __bf16* xb     = (__bf16*)(ws + off); off += (size_t)Bv * Tv * Cv * 2;       // 16 MB
  __bf16* wqkvb  = (__bf16*)(ws + off); off += (size_t)3 * Cv * Cv * 2;        //  6 MB
  __bf16* wprojb = (__bf16*)(ws + off); off += (size_t)Cv * Cv * 2;            //  2 MB
  __bf16* qb     = (__bf16*)(ws + off); off += (size_t)Bv * Tv * Cv * 2;       // 16 MB
  __bf16* kb     = (__bf16*)(ws + off); off += (size_t)Bv * Tv * Cv * 2;       // 16 MB
  __bf16* vT     = (__bf16*)(ws + off); off += (size_t)Bv * Tv * Cv * 2;       // 16 MB
  __bf16* ob     = (__bf16*)(ws + off); off += (size_t)Bv * Tv * Cv * 2;       // 16 MB

  // fp32 -> bf16 conversions
  {
    int n4 = (Bv * Tv * Cv) / 4;
    int blocks = (n4 + 255) / 256; if (blocks > 8192) blocks = 8192;
    cvt_f32_bf16<<<blocks, 256, 0, stream>>>(x, xb, n4);
  }
  {
    int n4 = (3 * Cv * Cv) / 4;
    int blocks = (n4 + 255) / 256; if (blocks > 8192) blocks = 8192;
    cvt_f32_bf16<<<blocks, 256, 0, stream>>>(wqkv, wqkvb, n4);
  }
  {
    int n4 = (Cv * Cv) / 4;
    int blocks = (n4 + 255) / 256; if (blocks > 8192) blocks = 8192;
    cvt_f32_bf16<<<blocks, 256, 0, stream>>>(wproj, wprojb, n4);
  }

  // QKV projection (M=8192, N=3072, K=1024)
  qkv_gemm<<<dim3((Bv * Tv) / 256, (3 * Cv) / 128), 256, 0, stream>>>(
      xb, wqkvb, qb, kb, vT);

  // Flash attention: B*H*(T/32/8) blocks of 8 waves
  flash_attn<<<Bv * Hv * 8, 256, 0, stream>>>(qb, kb, vT, ob);

  // Output projection (M=8192, N=1024, K=1024)
  proj_gemm<<<dim3((Bv * Tv) / 256, Cv / 128), 256, 0, stream>>>(ob, wprojb, out);
}
